// LoFormer_3633542332830
// MI455X (gfx1250) — matrix operand order repaired and checked by reading, not compiled, not run on hardware
//
#include <hip/hip_runtime.h>
#include <math.h>

// ---------------------------------------------------------------------------
// Types / bf16 helpers
// ---------------------------------------------------------------------------
typedef unsigned short u16;
typedef unsigned int   u32;
typedef unsigned long long u64;
typedef __attribute__((ext_vector_type(16))) __bf16 bf16x16;
typedef __attribute__((ext_vector_type(8)))  float  f32x8;

union FragB16 { bf16x16 v; uint4 q[2]; u32 d[8]; };
union U4H8    { uint4 q; u16 h[8]; };

static __device__ __forceinline__ float bf2f(u16 u) {
  union { u32 i; float f; } v; v.i = ((u32)u) << 16; return v.f;
}
static __device__ __forceinline__ u16 f2bfu(float f) {
  union { float f; u32 i; } v; v.f = f;
  u32 r = v.i + 0x7fffu + ((v.i >> 16) & 1u);   // round-to-nearest-even
  return (u16)(r >> 16);
}

static __device__ __forceinline__ f32x8 wmma_bf16(bf16x16 a, bf16x16 b, f32x8 c) {
  // D = A(16x32) * B(32x16) + C ; fp32 accumulate
  return __builtin_amdgcn_wmma_f32_16x16x32_bf16(false, a, false, b, (short)0, c,
                                                 false, false);
}

// ---------------------------------------------------------------------------
// Packed fragment loads (no per-element conversion).
//  A 16x32 bf16: lane<16 -> row=lane,     K {0..7,16..23}
//                lane>=16 -> row=lane-16, K {8..15,24..31}
//    -> dwords 0..3 = K kb..kb+7 (16B run), dwords 4..7 = kb+16..kb+23 (16B run)
//  B 32x16 bf16: lane<16 -> col=lane, K {0..7,16..23} ; lane>=16 -> col, K {8..15,24..31}
//    -> dwords 0..3 = K kb..kb+7 (16B run), dwords 4..7 = kb+16..kb+23 (16B run)
//  Requirements: base 16B aligned, ld (u16 units) multiple of 8, k0 mult of 32.
// ---------------------------------------------------------------------------
static __device__ __forceinline__ bf16x16 ldA_pk(const u16* m, int ld, int m0, int k0) {
  const int lane = threadIdx.x & 31;
  const int row = m0 + (lane & 15);
  const int kb  = k0 + ((lane >> 4) << 3);
  const u16* p = m + row * ld + kb;
  FragB16 f;
  f.q[0] = *(const uint4*)(p);
  f.q[1] = *(const uint4*)(p + 16);
  return f.v;
}
static __device__ __forceinline__ bf16x16 ldB_pk(const u16* m, int ld, int n0, int k0) {
  const int lane = threadIdx.x & 31;
  const int n  = n0 + (lane & 15);
  const int kb = k0 + ((lane >> 4) << 3);
  const u16* p = m + n * ld + kb;
  FragB16 f;
  f.q[0] = *(const uint4*)(p);
  f.q[1] = *(const uint4*)(p + 16);
  return f.v;
}

// ---------------------------------------------------------------------------
// Kernel 0: fp32 -> bf16 weight conversion (runs once per launch, tiny)
// ---------------------------------------------------------------------------
__global__ __launch_bounds__(256) void cvt_bf16(const float* __restrict__ in,
                                                u16* __restrict__ out, int n) {
  int i = blockIdx.x * 256 + threadIdx.x;
  if (i < n) out[i] = f2bfu(in[i]);
}

// ---------------------------------------------------------------------------
// Kernels 1 & 4: 1x1 conv as GEMM  out[b,o,hw] = sum_c W[o,c]*in[b,c,hw]
//  4 waves/block; wave -> 16(M) x 64(N); K staged 32-wide into LDS (bf16,
//  pixel-major ld=40 so B frags are two ds_load_b128). A frags: two
//  global_load_b128 from pre-converted bf16 weights.
// ---------------------------------------------------------------------------
template <bool IN_BF16, bool OUT_F32>
__global__ __launch_bounds__(128) void gemm1x1(const void* __restrict__ xin,
                                               const u16* __restrict__ Wbf,
                                               const float* __restrict__ bias,
                                               void* __restrict__ out,
                                               int OC, int K, int HW) {
  __shared__ __align__(16) u16 xt[64 * 40];  // [pixel][k], ld=40 (80B rows)
  const int b    = blockIdx.z;
  const int px0  = blockIdx.x * 64;
  const int wv   = threadIdx.x >> 5;
  const int lane = threadIdx.x & 31;
  const int tid  = (int)threadIdx.x;
  const int mrow = blockIdx.y * 64 + wv * 16;

  f32x8 acc[4];
#pragma unroll
  for (int nt = 0; nt < 4; ++nt)
#pragma unroll
    for (int e = 0; e < 8; ++e) acc[nt][e] = 0.f;

  for (int kc = 0; kc < K; kc += 32) {
    if (IN_BF16) {
      // raw bf16 copy: 256 x b128 chunks (8 halfwords each)
      for (int g = tid; g < 256; g += 128) {
        int i = g >> 3, j8 = (g & 7) * 8;
        U4H8 r;
        r.q = *(const uint4*)((const u16*)xin + (size_t)(b * K + kc + i) * HW + px0 + j8);
#pragma unroll
        for (int t = 0; t < 8; ++t) xt[(j8 + t) * 40 + i] = r.h[t];
      }
    } else {
      // fp32 load (b128) + convert once
      for (int g = tid; g < 512; g += 128) {
        int i = g >> 4, j4 = (g & 15) * 4;
        const float4 f4 =
            *(const float4*)((const float*)xin + (size_t)(b * K + kc + i) * HW + px0 + j4);
        xt[(j4 + 0) * 40 + i] = f2bfu(f4.x);
        xt[(j4 + 1) * 40 + i] = f2bfu(f4.y);
        xt[(j4 + 2) * 40 + i] = f2bfu(f4.z);
        xt[(j4 + 3) * 40 + i] = f2bfu(f4.w);
        if (kc + 32 < K)
          __builtin_prefetch((const float*)xin + (size_t)(b * K + kc + 32 + i) * HW + px0 + j4,
                             0, 3);
      }
    }
    __syncthreads();
    bf16x16 a = ldA_pk(Wbf, K, mrow, kc);
#pragma unroll
    for (int nt = 0; nt < 4; ++nt) {
      bf16x16 bb = ldB_pk(xt, 40, nt * 16, 0);
      acc[nt] = wmma_bf16(a, bb, acc[nt]);
    }
    __syncthreads();
  }

  const int hi = (lane >> 4) * 8;
#pragma unroll
  for (int nt = 0; nt < 4; ++nt)
#pragma unroll
    for (int r = 0; r < 8; ++r) {
      int m = mrow + r + hi;
      size_t idx = (size_t)(b * OC + m) * HW + px0 + nt * 16 + (lane & 15);
      float v = acc[nt][r];
      if (OUT_F32) ((float*)out)[idx] = v + bias[m];
      else         ((u16*)out)[idx]   = f2bfu(v);
    }
}

// ---------------------------------------------------------------------------
// Kernel 2: depthwise 3x3 (cross-correlation, zero pad) + scatter into the
// windowed layout qkv_w[z][win][head][c][x]  (bf16)
// ---------------------------------------------------------------------------
__global__ __launch_bounds__(256) void dwconv3x3(const u16* __restrict__ in,
                                                 const float* __restrict__ wdw,
                                                 u16* __restrict__ outw) {
  size_t tid = (size_t)blockIdx.x * 256 + threadIdx.x;
  int hw   = (int)(tid & 65535);
  int rest = (int)(tid >> 16);
  int o = rest % 576;
  int b = rest / 576;
  int h = hw >> 8, w = hw & 255;

  const u16* ip = in + ((size_t)(b * 576 + o) << 16);
  const float* wp = wdw + o * 9;
  float acc = 0.f;
#pragma unroll
  for (int dy = -1; dy <= 1; ++dy)
#pragma unroll
    for (int dx = -1; dx <= 1; ++dx) {
      int hh = h + dy, ww = w + dx;
      if (hh >= 0 && hh < 256 && ww >= 0 && ww < 256)
        acc += wp[(dy + 1) * 3 + (dx + 1)] * bf2f(ip[hh * 256 + ww]);
    }
  int z = o / 192, ch = o % 192, head = ch >> 5, c = ch & 31;
  int n = (b << 10) + ((h >> 3) << 5) + (w >> 3);
  int x = ((h & 7) << 3) + (w & 7);
  size_t dst = ((size_t)((z * 2048 + n) * 6 + head) << 11) + (c << 6) + x;
  outw[dst] = f2bfu(acc);
}

// ---------------------------------------------------------------------------
// Kernel 3: windowed channel attention; one wave = one (window, head).
//  q,k normalized in place in LDS; V kept twice (row-major + transposed) so
//  every WMMA operand is a packed b128 LDS/global load.
// ---------------------------------------------------------------------------
#define NW 2
__global__ __launch_bounds__(64) void win_attn(const u16* __restrict__ qkvw,
                                               const float* __restrict__ temp,
                                               const u16* __restrict__ mlpbf,
                                               const float* __restrict__ mlpb,
                                               u16* __restrict__ aout) {
  __shared__ __align__(16) u16   qs[NW][32 * 40];
  __shared__ __align__(16) u16   ks[NW][32 * 40];
  __shared__ __align__(16) u16   vs[NW][32 * 40];
  __shared__ __align__(16) u16   vt[NW][64 * 40];  // V^T : [x][d]
  __shared__ __align__(16) u16   pb[NW][32 * 40];
  __shared__ float               sb[NW][32 * 36];

  const int wv   = threadIdx.x >> 5;
  const int lane = threadIdx.x & 31;
  const int unit = blockIdx.x * NW + wv;       // = win*6 + head
  const int nwin = unit / 6, head = unit % 6;
  const size_t base = (size_t)unit * 2048;
  const size_t zs   = 25165824ull;             // 2048*6*2048

  // ---- stage q,k,v (bf16 b128 chunks; ld=40 keeps rows 16B aligned) ----
#pragma unroll
  for (int it = 0; it < 8; ++it) {
    int cidx = it * 32 + lane;                 // 256 chunks of 8 halfwords
    int row = cidx >> 3;
    int off = (cidx & 7) * 8;
    uint4 a0 = *(const uint4*)(qkvw + base + row * 64 + off);
    uint4 a1 = *(const uint4*)(qkvw + zs + base + row * 64 + off);
    uint4 a2 = *(const uint4*)(qkvw + 2 * zs + base + row * 64 + off);
    *(uint4*)(&qs[wv][row * 40 + off]) = a0;
    *(uint4*)(&ks[wv][row * 40 + off]) = a1;
    *(uint4*)(&vs[wv][row * 40 + off]) = a2;
  }
  __syncthreads();

  // ---- normalize q,k rows in place; build V^T (lane r owns row r) ----
  {
    u16* qr = &qs[wv][lane * 40];
    u16* kr = &ks[wv][lane * 40];
    const u16* vr = &vs[wv][lane * 40];
    float sq = 0.f, sk = 0.f;
#pragma unroll
    for (int xx = 0; xx < 64; ++xx) {
      float a = bf2f(qr[xx]); sq += a * a;
      float c = bf2f(kr[xx]); sk += c * c;
    }
    float qsc = 1.f / fmaxf(sqrtf(sq), 1e-12f);
    float ksc = 1.f / fmaxf(sqrtf(sk), 1e-12f);
#pragma unroll
    for (int xx = 0; xx < 64; ++xx) {
      qr[xx] = f2bfu(bf2f(qr[xx]) * qsc);
      kr[xx] = f2bfu(bf2f(kr[xx]) * ksc);
      vt[wv][xx * 40 + lane] = vr[xx];
    }
  }
  __syncthreads();

  const int hi = (lane >> 4) * 8;

  // ---- S(32x32) = qn * kn^T ----
#pragma unroll
  for (int mt = 0; mt < 2; ++mt)
#pragma unroll
    for (int nt = 0; nt < 2; ++nt) {
      f32x8 acc;
#pragma unroll
      for (int e = 0; e < 8; ++e) acc[e] = 0.f;
#pragma unroll
      for (int kt = 0; kt < 2; ++kt) {
        bf16x16 a  = ldA_pk(qs[wv], 40, mt * 16, kt * 32);
        bf16x16 bb = ldB_pk(ks[wv], 40, nt * 16, kt * 32);
        acc = wmma_bf16(a, bb, acc);
      }
#pragma unroll
      for (int r = 0; r < 8; ++r)
        sb[wv][(mt * 16 + r + hi) * 36 + nt * 16 + (lane & 15)] = acc[r];
    }
  __syncthreads();

  // ---- softmax over d (row = lane), temperature on logits ----
  {
    float tpr = temp[head];
    float mx = -3.0e38f;
    for (int d = 0; d < 32; ++d) mx = fmaxf(mx, sb[wv][lane * 36 + d] * tpr);
    float s = 0.f;
    for (int d = 0; d < 32; ++d) s += __expf(sb[wv][lane * 36 + d] * tpr - mx);
    float inv = 1.f / s;
    for (int d = 0; d < 32; ++d)
      pb[wv][lane * 40 + d] = f2bfu(__expf(sb[wv][lane * 36 + d] * tpr - mx) * inv);
  }
  __syncthreads();

  // ---- O = P*V (K=32) and G = V*mlp^T (K=64) ----
  f32x8 accO[2][4], accG[2][4];
#pragma unroll
  for (int mt = 0; mt < 2; ++mt)
#pragma unroll
    for (int nt = 0; nt < 4; ++nt)
#pragma unroll
      for (int e = 0; e < 8; ++e) { accO[mt][nt][e] = 0.f; accG[mt][nt][e] = 0.f; }

  bf16x16 aP[2], aV0[2], aV1[2];
#pragma unroll
  for (int mt = 0; mt < 2; ++mt) {
    aP[mt]  = ldA_pk(pb[wv], 40, mt * 16, 0);
    aV0[mt] = ldA_pk(vs[wv], 40, mt * 16, 0);
    aV1[mt] = ldA_pk(vs[wv], 40, mt * 16, 32);
  }
#pragma unroll
  for (int nt = 0; nt < 4; ++nt) {
    bf16x16 bV  = ldB_pk(vt[wv], 40, nt * 16, 0);       // B[K=d][N=x] = V^T[x][d]
    bf16x16 bM0 = ldB_pk(mlpbf, 64, nt * 16, 0);        // B[K=x][N=y] = mlp[y][x]
    bf16x16 bM1 = ldB_pk(mlpbf, 64, nt * 16, 32);
#pragma unroll
    for (int mt = 0; mt < 2; ++mt) {
      accO[mt][nt] = wmma_bf16(aP[mt],  bV,  accO[mt][nt]);
      accG[mt][nt] = wmma_bf16(aV0[mt], bM0, accG[mt][nt]);
      accG[mt][nt] = wmma_bf16(aV1[mt], bM1, accG[mt][nt]);
    }
  }

  // ---- out = O * GELU(G + b), scatter back to NCHW (bf16) ----
  const int b = nwin >> 10, rem = nwin & 1023;
  const int h1 = rem >> 5, w1 = rem & 31;
#pragma unroll
  for (int mt = 0; mt < 2; ++mt)
#pragma unroll
    for (int nt = 0; nt < 4; ++nt)
#pragma unroll
      for (int r = 0; r < 8; ++r) {
        int m  = mt * 16 + r + hi;
        int xx = nt * 16 + (lane & 15);
        float g = accG[mt][nt][r] + mlpb[xx];
        float gel = 0.5f * g * (1.f + erff(g * 0.70710678118f));
        float val = accO[mt][nt][r] * gel;
        int hh = h1 * 8 + (xx >> 3), ww = w1 * 8 + (xx & 7);
        aout[((size_t)(b * 192 + head * 32 + m) << 16) + hh * 256 + ww] = f2bfu(val);
      }
}

// ---------------------------------------------------------------------------
// Launch
// ---------------------------------------------------------------------------
extern "C" void kernel_launch(void* const* d_in, const int* in_sizes, int n_in,
                              void* d_out, int out_size, void* d_ws, size_t ws_size,
                              hipStream_t stream) {
  (void)in_sizes; (void)n_in; (void)out_size; (void)ws_size;
  const float* x    = (const float*)d_in[0];
  const float* wqkv = (const float*)d_in[1];
  const float* wdw  = (const float*)d_in[2];
  const float* temp = (const float*)d_in[3];
  const float* mlpw = (const float*)d_in[4];
  const float* mlpb = (const float*)d_in[5];
  const float* pw   = (const float*)d_in[6];
  const float* pb   = (const float*)d_in[7];
  float* out = (float*)d_out;

  // workspace layout (all regions 16B aligned)
  u16* qkv_lin = (u16*)d_ws;                             // 2*576*65536   bf16
  u16* qkv_w   = (u16*)((char*)d_ws + 150994944ull);     // 3*2048*6*2048 bf16
  u16* attn_o  = (u16*)((char*)d_ws + 301989888ull);     // 2*192*65536   bf16
  u16* wqkv_bf = (u16*)((char*)d_ws + 352321536ull);     // 576*192
  u16* pw_bf   = (u16*)((char*)d_ws + 352542720ull);     // 192*192
  u16* mlp_bf  = (u16*)((char*)d_ws + 352616448ull);     // 64*64

  // 0) one-time weight conversions (bf16)
  cvt_bf16<<<dim3((110592 + 255) / 256), 256, 0, stream>>>(wqkv, wqkv_bf, 110592);
  cvt_bf16<<<dim3((36864 + 255) / 256), 256, 0, stream>>>(pw, pw_bf, 36864);
  cvt_bf16<<<dim3((4096 + 255) / 256), 256, 0, stream>>>(mlpw, mlp_bf, 4096);

  // 1) qkv = W_qkv (576x192) * x   -> bf16 NCHW
  gemm1x1<false, false><<<dim3(1024, 9, 2), 128, 0, stream>>>(
      (const void*)x, wqkv_bf, nullptr, (void*)qkv_lin, 576, 192, 65536);

  // 2) depthwise 3x3 + windowed re-layout
  dwconv3x3<<<dim3(294912), 256, 0, stream>>>(qkv_lin, wdw, qkv_w);

  // 3) windowed channel attention + MLP gate
  win_attn<<<dim3(6144), 64, 0, stream>>>(qkv_w, temp, mlp_bf, mlpb, attn_o);

  // 4) projection: out = proj_w (192x192) * attn + proj_b  -> fp32
  gemm1x1<true, true><<<dim3(1024, 3, 2), 128, 0, stream>>>(
      (const void*)attn_o, pw_bf, pb, (void*)out, 192, 192, 65536);
}
